// MatryoshkaSAE_76295799046879
// MI455X (gfx1250) — matrix operand order repaired and checked by reading, not compiled, not run on hardware
//
#include <hip/hip_runtime.h>
#include <hip/hip_bf16.h>
#include <stdint.h>

// ---------------------------------------------------------------------------
// MatryoshkaSAE forward for MI455X (gfx1250), wave32 + WMMA bf16 path,
// async global->LDS double-buffered B tiles (ASYNCcnt).
// ---------------------------------------------------------------------------

typedef __attribute__((ext_vector_type(8)))  float  v8f;
typedef __attribute__((ext_vector_type(16))) __bf16 v16bf;

#define INPUT_DIM 1536
#define BATCH     1024
#define MAX_DICT  32768
#define KT_ENC    48      // 1536 / 32
#define KT_DEC_MAX 1024   // 32768 / 32

// ---- helpers --------------------------------------------------------------

__device__ __forceinline__ unsigned short f32_to_bf16_bits(float f) {
  unsigned u = __float_as_uint(f);
  unsigned r = u + 0x7FFFu + ((u >> 16) & 1u);   // round-to-nearest-even
  return (unsigned short)(r >> 16);
}

// monotone map: float -> uint32 preserving total order (descending = larger key)
__device__ __forceinline__ unsigned fmap(float v) {
  unsigned u = __float_as_uint(v);
  return (u & 0x80000000u) ? ~u : (u | 0x80000000u);
}

// async copy of one 32-byte per-lane chunk: global -> LDS.
// INST_OFFSET applies to both LDS and global address (ISA 08, §4.4).
// Passing the *address* of the LDS destination (ptrtoint) captures the shared
// array, so the "memory" clobber legally covers it and the later ds_loads
// cannot be folded away. Low 32 bits of the generic address == LDS offset.
__device__ __forceinline__ void async_b128x2(const unsigned short* lds_dst,
                                             const void* gaddr) {
  unsigned loff = (unsigned)(unsigned long long)lds_dst;
  asm volatile("global_load_async_to_lds_b128 %0, %1, off"
               :: "v"(loff), "v"(gaddr) : "memory");
  asm volatile("global_load_async_to_lds_b128 %0, %1, off offset:16"
               :: "v"(loff), "v"(gaddr) : "memory");
}

// ---- pack kernels: build fragment-major bf16 buffers ----------------------
// A-fragment (16x32 bf16, wave32): lane l (half = l>>4) holds row mt*16+(l&15),
//   element j -> k = kt*32 + (j&7) + 8*half + 16*(j>>3)   (per ISA 7.12.2)
// B-fragment (32x16 bf16): lane l holds col nt*16+(l&15),
//   element j -> k = kt*32 + (j&7) + 8*(l>>4) + 16*(j>>3)

__global__ void pack_xb_kernel(const float* __restrict__ x,
                               const float* __restrict__ bias,
                               unsigned short* __restrict__ apack) {
  int i = blockIdx.x * blockDim.x + threadIdx.x;
  if (i >= BATCH * INPUT_DIM) return;
  int j = i & 15, l = (i >> 4) & 31, rest = i >> 9;
  int kt = rest % KT_ENC, mt = rest / KT_ENC;
  int half = l >> 4;
  int row = mt * 16 + (l & 15);
  int k = kt * 32 + (j & 7) + 8 * half + 16 * (j >> 3);
  apack[i] = f32_to_bf16_bits(x[row * INPUT_DIM + k] - bias[k]);
}

// encoder B:  B[k][n] = W[n][k]  (reads are contiguous along W rows)
__global__ void pack_wenc_kernel(const float* __restrict__ W,
                                 unsigned short* __restrict__ bpack) {
  long i = (long)blockIdx.x * blockDim.x + threadIdx.x;
  if (i >= (long)MAX_DICT * INPUT_DIM) return;
  int j = (int)(i & 15), l = (int)((i >> 4) & 31);
  long rest = i >> 9;
  int kt = (int)(rest % KT_ENC), nt = (int)(rest / KT_ENC);
  int n = nt * 16 + (l & 15);
  int k = kt * 32 + (j & 7) + 8 * (l >> 4) + 16 * (j >> 3);
  bpack[i] = f32_to_bf16_bits(W[(long)n * INPUT_DIM + k]);
}

// decoder B:  B[k][n] = W[k][n]  (k over dict dim, n over 1536)
__global__ void pack_wdec_kernel(const float* __restrict__ W,
                                 unsigned short* __restrict__ bpack) {
  long i = (long)blockIdx.x * blockDim.x + threadIdx.x;
  if (i >= (long)MAX_DICT * INPUT_DIM) return;
  int j = (int)(i & 15), l = (int)((i >> 4) & 31);
  long rest = i >> 9;
  int kt = (int)(rest & (KT_DEC_MAX - 1));
  int nt = (int)(rest >> 10);                 // 0..95
  int n = nt * 16 + (l & 15);
  int k = kt * 32 + (j & 7) + 8 * (l >> 4) + 16 * (j >> 3);
  bpack[i] = f32_to_bf16_bits(W[(long)k * INPUT_DIM + n]);
}

__global__ void pack_sparse_kernel(const float* __restrict__ sp, int s, int ktshift,
                                   unsigned short* __restrict__ apack) {
  long i = (long)blockIdx.x * blockDim.x + threadIdx.x;
  if (i >= (long)BATCH * s) return;
  int j = (int)(i & 15), l = (int)((i >> 4) & 31);
  long rest = i >> 9;
  int ktmask = (1 << ktshift) - 1;
  int kt = (int)(rest & ktmask);
  int mt = (int)(rest >> ktshift);
  int half = l >> 4;
  int row = mt * 16 + (l & 15);
  int k = kt * 32 + (j & 7) + 8 * half + 16 * (j >> 3);
  apack[i] = f32_to_bf16_bits(sp[(long)row * s + k]);
}

// ---- encoder GEMM:  latents = xb @ W^T + b_enc ----------------------------
// block = 8 waves, 128(M) x 128(N) tile; B tile double-buffered in LDS via
// async global->LDS loads (ASYNCcnt), 8 WMMA per K-step per wave.

__global__ void __launch_bounds__(256)
enc_gemm_kernel(const unsigned short* __restrict__ apack,
                const unsigned short* __restrict__ bpack,
                const float* __restrict__ b_enc,
                float* __restrict__ lat) {
  __shared__ unsigned short lb[2][8 * 32 * 16];   // 2 x 8 KB B tiles
  const int lane = threadIdx.x & 31;
  const int wave = threadIdx.x >> 5;
  const int bn = blockIdx.x;          // 0..255
  const int bm = blockIdx.y;          // 0..7
  const int mt = bm * 8 + wave;
  const int nt0 = bn * 8;

  v8f acc[8];
  #pragma unroll
  for (int j = 0; j < 8; ++j)
    #pragma unroll
    for (int r = 0; r < 8; ++r) acc[j][r] = 0.f;

  const unsigned short* abase = apack + ((long)mt * KT_ENC) * 512 + lane * 16;
  const int chunk = (wave * 32 + lane) * 16;      // ushort index of this lane's chunk

  // prologue: fetch tile for kt=0 into buffer 0
  async_b128x2(&lb[0][chunk],
               bpack + ((long)((nt0 + wave) * KT_ENC + 0)) * 512 + lane * 16);

  for (int kt = 0; kt < KT_ENC; ++kt) {
    const int cur = kt & 1;
    if (kt + 1 < KT_ENC) {
      async_b128x2(&lb[cur ^ 1][chunk],
                   bpack + ((long)((nt0 + wave) * KT_ENC + kt + 1)) * 512 + lane * 16);
      asm volatile("s_wait_asynccnt 0x2" ::: "memory");  // older tile done (in-order)
    } else {
      asm volatile("s_wait_asynccnt 0x0" ::: "memory");
    }
    v16bf a = *reinterpret_cast<const v16bf*>(abase + (long)kt * 512);
    __syncthreads();                    // tile `cur` complete in every wave
    #pragma unroll
    for (int j = 0; j < 8; ++j) {
      v16bf b = *reinterpret_cast<const v16bf*>(&lb[cur][(j * 32 + lane) * 16]);
      acc[j] = __builtin_amdgcn_wmma_f32_16x16x32_bf16(
          false, a, false, b, (short)0, acc[j], false, false);
    }
    __syncthreads();                    // all reads of tile `cur` finished
  }

  // C/D layout: VGPR r, lane l -> m = r + 8*(l>>4), n = l&15
  const int rowb = mt * 16 + ((lane >> 4) << 3);
  const int cl = lane & 15;
  #pragma unroll
  for (int j = 0; j < 8; ++j) {
    int col = (nt0 + j) * 16 + cl;
    float be = b_enc[col];
    #pragma unroll
    for (int r = 0; r < 8; ++r)
      lat[(long)(rowb + r) * MAX_DICT + col] = acc[j][r] + be;
  }
}

// ---- decoder GEMM (split-K, atomics into bias-pre-initialized recon) ------

__global__ void __launch_bounds__(256)
dec_gemm_kernel(const unsigned short* __restrict__ apack,
                const unsigned short* __restrict__ bpack,
                int KtS, int sliceLen,
                float* __restrict__ recon) {
  __shared__ unsigned short lb[2][8 * 32 * 16];
  const int lane = threadIdx.x & 31;
  const int wave = threadIdx.x >> 5;
  const int bn = blockIdx.x;          // 0..11  (N = 1536 -> 12 blocks of 128)
  const int bm = blockIdx.y;          // 0..7
  const int sl = blockIdx.z;          // K slice
  const int mt = bm * 8 + wave;
  const int nt0 = bn * 8;

  v8f acc[8];
  #pragma unroll
  for (int j = 0; j < 8; ++j)
    #pragma unroll
    for (int r = 0; r < 8; ++r) acc[j][r] = 0.f;

  const int kt0 = sl * sliceLen;
  const int chunk = (wave * 32 + lane) * 16;

  async_b128x2(&lb[0][chunk],
               bpack + ((long)(nt0 + wave) * KT_DEC_MAX + kt0) * 512 + lane * 16);

  for (int kk = 0; kk < sliceLen; ++kk) {
    const int kt = kt0 + kk;
    const int cur = kk & 1;
    if (kk + 1 < sliceLen) {
      async_b128x2(&lb[cur ^ 1][chunk],
                   bpack + ((long)(nt0 + wave) * KT_DEC_MAX + kt + 1) * 512 + lane * 16);
      asm volatile("s_wait_asynccnt 0x2" ::: "memory");
    } else {
      asm volatile("s_wait_asynccnt 0x0" ::: "memory");
    }
    v16bf a = *reinterpret_cast<const v16bf*>(
        apack + ((long)mt * KtS + kt) * 512 + lane * 16);
    __syncthreads();
    #pragma unroll
    for (int j = 0; j < 8; ++j) {
      v16bf b = *reinterpret_cast<const v16bf*>(&lb[cur][(j * 32 + lane) * 16]);
      acc[j] = __builtin_amdgcn_wmma_f32_16x16x32_bf16(
          false, a, false, b, (short)0, acc[j], false, false);
    }
    __syncthreads();
  }

  const int rowb = mt * 16 + ((lane >> 4) << 3);
  const int cl = lane & 15;
  #pragma unroll
  for (int j = 0; j < 8; ++j) {
    int col = (nt0 + j) * 16 + cl;
    #pragma unroll
    for (int r = 0; r < 8; ++r)
      atomicAdd(&recon[(long)(rowb + r) * INPUT_DIM + col], acc[j][r]);
  }
}

// ---- BatchTopK via two-level 12-bit radix histogram threshold -------------

__global__ void zero_u32_kernel(unsigned* __restrict__ p, int n) {
  int i = blockIdx.x * blockDim.x + threadIdx.x;
  if (i < n) p[i] = 0u;
}

// mode 0: histogram of top-12 key bits.  mode 1: next-12 bits within bucket B1.
__global__ void hist_kernel(const float* __restrict__ lat, int sshift, long total,
                            unsigned* __restrict__ hist,
                            const int* __restrict__ ctrl, int mode) {
  __shared__ unsigned lh[4096];
  for (int i = threadIdx.x; i < 4096; i += blockDim.x) lh[i] = 0u;
  unsigned b1 = (mode != 0) ? (unsigned)ctrl[0] : 0u;
  __syncthreads();
  const int smask = (1 << sshift) - 1;
  const long stride = (long)gridDim.x * blockDim.x;
  for (long i = (long)blockIdx.x * blockDim.x + threadIdx.x; i < total; i += stride) {
    int m = (int)(i >> sshift), c = (int)(i & smask);
    float v = lat[(long)m * MAX_DICT + c];
    unsigned u = fmap(v);
    if (mode == 0) {
      atomicAdd(&lh[u >> 20], 1u);
    } else if ((u >> 20) == b1) {
      atomicAdd(&lh[(u >> 8) & 0xFFFu], 1u);
    }
  }
  __syncthreads();
  for (int i = threadIdx.x; i < 4096; i += blockDim.x) {
    unsigned c = lh[i];
    if (c) atomicAdd(&hist[i], c);
  }
}

// descending suffix scan to locate threshold bucket (single thread; 4096 iters)
__global__ void scan_kernel(const unsigned* __restrict__ hist, int targetN,
                            int* __restrict__ ctrl, int phase) {
  if (threadIdx.x != 0 || blockIdx.x != 0) return;
  long N = (phase == 0) ? (long)targetN : (long)ctrl[1];
  long acc = 0;
  int b = 4095;
  for (; b > 0; --b) {
    long h = hist[b];
    if (acc + h >= N) break;
    acc += h;
  }
  if (phase == 0) {
    ctrl[0] = b;                 // B1
    ctrl[1] = (int)(N - acc);    // remaining needed from bucket B1
  } else {
    ctrl[2] = (int)(((unsigned)ctrl[0] << 12) | (unsigned)b);  // 24-bit key T
    ctrl[3] = (int)(N - acc);    // tie budget at key == T
  }
}

__global__ void scatter_kernel(const float* __restrict__ lat, int sshift, long total,
                               int* __restrict__ ctrl, float* __restrict__ outSp) {
  const unsigned T = (unsigned)ctrl[2];
  const int smask = (1 << sshift) - 1;
  const long stride = (long)gridDim.x * blockDim.x;
  for (long i = (long)blockIdx.x * blockDim.x + threadIdx.x; i < total; i += stride) {
    int m = (int)(i >> sshift), c = (int)(i & smask);
    float v = lat[(long)m * MAX_DICT + c];
    unsigned k24 = fmap(v) >> 8;
    float o = 0.f;
    if (k24 > T) {
      o = v;
    } else if (k24 == T) {
      int old = atomicAdd(&ctrl[3], -1);
      if (old > 0) o = v;
    }
    outSp[i] = o;   // writes entire sparse region (also serves as zero-init)
  }
}

// ---- small epilogue helpers ----------------------------------------------

__global__ void bias_init_kernel(const float* __restrict__ bias,
                                 float* __restrict__ out) {
  long i = (long)blockIdx.x * blockDim.x + threadIdx.x;
  if (i >= 3L * BATCH * INPUT_DIM) return;
  int col = (int)(i % INPUT_DIM);
  out[i] = bias[col];
}

__global__ void copy_kernel(const float* __restrict__ src,
                            float* __restrict__ dst, long n) {
  long i = (long)blockIdx.x * blockDim.x + threadIdx.x;
  if (i < n) dst[i] = src[i];
}

// ---------------------------------------------------------------------------

extern "C" void kernel_launch(void* const* d_in, const int* in_sizes, int n_in,
                              void* d_out, int out_size, void* d_ws, size_t ws_size,
                              hipStream_t stream) {
  const float* x     = (const float*)d_in[0];
  const float* W     = (const float*)d_in[1];
  const float* b_enc = (const float*)d_in[2];
  const float* bias  = (const float*)d_in[3];
  float* out = (float*)d_out;
  char*  ws  = (char*)d_ws;

  // workspace layout (bytes, 256-aligned)
  const size_t OFF_CTRL = 0;                         // 4 ints
  const size_t OFF_HIST = 256;                       // 2 x 4096 u32
  const size_t OFF_XB   = 33024;                     // 1024*1536 bf16
  const size_t OFF_WP   = OFF_XB  + 3145728;         // 32768*1536 bf16 (enc, then dec)
  const size_t OFF_LAT  = OFF_WP  + 100663296;       // 1024*32768 f32
  const size_t OFF_SP   = OFF_LAT + 134217728;       // 1024*32768 bf16

  int*            ctrl  = (int*)(ws + OFF_CTRL);
  unsigned*       hist1 = (unsigned*)(ws + OFF_HIST);
  unsigned*       hist2 = hist1 + 4096;
  unsigned short* xbp   = (unsigned short*)(ws + OFF_XB);
  unsigned short* wpack = (unsigned short*)(ws + OFF_WP);
  float*          lat   = (float*)(ws + OFF_LAT);
  unsigned short* spack = (unsigned short*)(ws + OFF_SP);

  const long RN = (long)BATCH * INPUT_DIM;           // 1,572,864
  float* finalR = out;
  float* recon  = out + RN;                          // recon0..2 contiguous
  float* sparse0 = out + 4 * RN;

  // 1) pack (x - bias) -> A fragments, W^T -> encoder B fragments
  pack_xb_kernel<<<(BATCH * INPUT_DIM + 255) / 256, 256, 0, stream>>>(x, bias, xbp);
  const long wn = (long)MAX_DICT * INPUT_DIM;
  pack_wenc_kernel<<<(unsigned)((wn + 255) / 256), 256, 0, stream>>>(W, wpack);

  // 2) encoder GEMM -> latents_pre (f32)
  dim3 eg(MAX_DICT / 128, BATCH / 128, 1);           // (256, 8)
  enc_gemm_kernel<<<eg, 256, 0, stream>>>(xbp, wpack, b_enc, lat);

  // 3) repack W for decoder (reuses wpack; stream order guarantees safety)
  pack_wdec_kernel<<<(unsigned)((wn + 255) / 256), 256, 0, stream>>>(W, wpack);

  // 4) recon buffers start at bias
  bias_init_kernel<<<(unsigned)((3 * RN + 255) / 256), 256, 0, stream>>>(bias, recon);

  const int sizes[3]  = {2048, 8192, 32768};
  const int sshift[3] = {11, 13, 15};
  const int nsel[3]   = {102 * 1024, 409 * 1024, 1638 * 1024};
  float* sparsePtr = sparse0;

  for (int si = 0; si < 3; ++si) {
    const int  s     = sizes[si];
    const long total = (long)BATCH * s;
    const int  KtS   = s / 32;
    const int  ktsh  = sshift[si] - 5;

    // BatchTopK threshold (two-level radix) + scatter
    zero_u32_kernel<<<32, 256, 0, stream>>>(hist1, 8192);
    hist_kernel<<<2048, 256, 0, stream>>>(lat, sshift[si], total, hist1, ctrl, 0);
    scan_kernel<<<1, 32, 0, stream>>>(hist1, nsel[si], ctrl, 0);
    hist_kernel<<<2048, 256, 0, stream>>>(lat, sshift[si], total, hist2, ctrl, 1);
    scan_kernel<<<1, 32, 0, stream>>>(hist2, 0, ctrl, 1);
    scatter_kernel<<<2048, 256, 0, stream>>>(lat, sshift[si], total, ctrl, sparsePtr);

    // decoder: sparse @ W[:s] (+bias via init), split-K over 8 slices
    pack_sparse_kernel<<<(unsigned)((total + 255) / 256), 256, 0, stream>>>(
        sparsePtr, s, ktsh, spack);
    dim3 dg(INPUT_DIM / 128, BATCH / 128, 8);        // (12, 8, 8)
    dec_gemm_kernel<<<dg, 256, 0, stream>>>(spack, wpack, KtS, KtS / 8,
                                            recon + (long)si * RN);
    sparsePtr += total;
  }

  // final_recon = recon2
  copy_kernel<<<(unsigned)((RN + 255) / 256), 256, 0, stream>>>(recon + 2 * RN, finalR, RN);
}
